// RwkvSelfAttention_70832600646182
// MI455X (gfx1250) — matrix operand order repaired and checked by reading, not compiled, not run on hardware
//
#include <hip/hip_runtime.h>
#include <cstdint>
#include <cstddef>

#define T_LEN 2048
#define D_DIM 2048
#define H_NUM 32
#define S_DIM 64
#define CHUNK 64

typedef __attribute__((ext_vector_type(16))) __bf16 v16bf;
typedef __attribute__((ext_vector_type(8)))  __bf16 v8bf;
typedef __attribute__((ext_vector_type(8)))  float  v8f;

// ---------------------------------------------------------------------------
// WMMA helpers (CDNA5: D = A(16x32 bf16) * B(32x16 bf16) + C(16x16 f32))
// ---------------------------------------------------------------------------
static __device__ __forceinline__ v8f wmma_bf16(v16bf a, v16bf b, v8f c) {
    return __builtin_amdgcn_wmma_f32_16x16x32_bf16(
        /*neg_a=*/false, a, /*neg_b=*/false, b,
        /*c_mod=*/(short)0, c, /*reuse_a=*/false, /*reuse_b=*/false);
}

// A-fragment (16x32, row-major source, row stride ld elements).
// A layout (matches B): lanes 0-15 hold row M=l16, K = kk..kk+15;
// lanes 16-31 hold row M=l16, K = kk+16..kk+31.
static __device__ __forceinline__ v16bf load_a_frag(const __bf16* base, int ld,
                                                    int kk, int l16, int half) {
    const __bf16* p = base + l16 * ld + kk + 16 * half;
    v8bf lo = *(const v8bf*)p;
    v8bf hi = *(const v8bf*)(p + 8);
    return __builtin_shufflevector(lo, hi, 0,1,2,3,4,5,6,7,8,9,10,11,12,13,14,15);
}

// B-fragment (32x16). Source is "column-major as rows": row n of `base` holds
// column n of B, K-contiguous. Lanes 0-15: K = kk..kk+15; lanes 16-31: kk+16..kk+31.
static __device__ __forceinline__ v16bf load_b_frag(const __bf16* base, int ld,
                                                    int kk, int l16, int half) {
    return *(const v16bf*)(base + l16 * ld + kk + 16 * half);
}

// ---------------------------------------------------------------------------
// CDNA5 async global -> LDS copy (ASYNCcnt path)
// ---------------------------------------------------------------------------
static __device__ __forceinline__ void async_load_b128(uint32_t lds_off, uint64_t gaddr) {
    asm volatile("global_load_async_to_lds_b128 %0, %1, off"
                 :: "v"(lds_off), "v"(gaddr) : "memory");
}
static __device__ __forceinline__ void wait_asynccnt0() {
    asm volatile("s_wait_asynccnt 0" ::: "memory");
}

// ---------------------------------------------------------------------------
// f32 -> bf16 bulk convert
// ---------------------------------------------------------------------------
__global__ __launch_bounds__(256) void cvt_f32_bf16_kernel(
    const float* __restrict__ src, __bf16* __restrict__ dst, int n) {
    int i = blockIdx.x * blockDim.x + threadIdx.x;
    if (i < n) dst[i] = (__bf16)src[i];
}

// ---------------------------------------------------------------------------
// Time-shift mixing: k_in/v_in/r_in = h*m + h_prev*(1-m), stored bf16
// ---------------------------------------------------------------------------
__global__ __launch_bounds__(256) void time_mix_kernel(
    const float* __restrict__ hidden,
    const float* __restrict__ tmk, const float* __restrict__ tmv,
    const float* __restrict__ tmr,
    __bf16* __restrict__ Kin, __bf16* __restrict__ Vin, __bf16* __restrict__ Rin) {
    int idx = blockIdx.x * blockDim.x + threadIdx.x;
    if (idx >= T_LEN * D_DIM) return;
    int t = idx >> 11;            // / D_DIM
    int d = idx & (D_DIM - 1);
    float h  = hidden[idx];
    float hp = (t == 0) ? 0.0f : hidden[idx - D_DIM];
    float mk = tmk[d], mv = tmv[d], mr = tmr[d];
    Kin[idx] = (__bf16)(h * mk + hp * (1.0f - mk));
    Vin[idx] = (__bf16)(h * mv + hp * (1.0f - mv));
    Rin[idx] = (__bf16)(h * mr + hp * (1.0f - mr));
}

// ---------------------------------------------------------------------------
// GEMM: C[M,N] = A[M,K] @ W[N,K]^T   (A,W bf16; C bf16 or f32)
// Block = 8 waves (256 thr), block tile 128x64, K staged in 64-wide LDS slabs
// filled by async global->LDS B128 copies. Each wave owns a 16x64 strip.
// ---------------------------------------------------------------------------
#define GEMM_BM 128
#define GEMM_BN 64
#define GEMM_BK 64

template <bool OUT_BF16>
__global__ __launch_bounds__(256) void gemm_wmma_kernel(
    const __bf16* __restrict__ A, const __bf16* __restrict__ W,
    void* __restrict__ Cout, int M, int N, int K) {
    __shared__ __attribute__((aligned(32))) __bf16 As[GEMM_BM * GEMM_BK]; // 16 KB
    __shared__ __attribute__((aligned(32))) __bf16 Bs[GEMM_BN * GEMM_BK]; //  8 KB

    const int tid  = threadIdx.x;
    const int lane = tid & 31;
    const int wave = tid >> 5;
    const int half = lane >> 4;
    const int l16  = lane & 15;
    const int rowB = blockIdx.x * GEMM_BM;          // block row base
    const int col0 = blockIdx.y * GEMM_BN;          // block col base
    const int row0 = rowB + wave * 16;              // wave's 16-row strip

    const __bf16* Ag = A + (size_t)rowB * K;
    const __bf16* Wg = W + (size_t)col0 * K;
    const uint32_t as_off = (uint32_t)(uintptr_t)(void*)As;
    const uint32_t bs_off = (uint32_t)(uintptr_t)(void*)Bs;

    v8f acc[4] = {};
    for (int kk = 0; kk < K; kk += GEMM_BK) {
        // ---- async stage A (128x64) and B (64x64) slabs into LDS ----
        // A: 128 rows * 8 16B-segments = 1024 ops; B: 64 * 8 = 512 ops.
        for (int o = tid; o < 1024; o += 256) {
            int r = o >> 3, seg = o & 7;
            async_load_b128(as_off + (uint32_t)(r * GEMM_BK + seg * 8) * 2,
                            (uint64_t)(uintptr_t)(Ag + (size_t)r * K + kk + seg * 8));
        }
        for (int o = tid; o < 512; o += 256) {
            int r = o >> 3, seg = o & 7;
            async_load_b128(bs_off + (uint32_t)(r * GEMM_BK + seg * 8) * 2,
                            (uint64_t)(uintptr_t)(Wg + (size_t)r * K + kk + seg * 8));
        }
        // prefetch next K slab (global_prefetch_b8) while this one computes
        if (kk + GEMM_BK < K) {
            __builtin_prefetch(Ag + (size_t)(tid >> 1) * K + kk + GEMM_BK + (tid & 1) * 32, 0, 0);
            __builtin_prefetch(Wg + (size_t)(tid & 63) * K + kk + GEMM_BK, 0, 0);
        }
        wait_asynccnt0();
        __syncthreads();

        // ---- compute out of LDS: 2 k-steps of 32, 4 N tiles ----
#pragma unroll
        for (int k2 = 0; k2 < GEMM_BK; k2 += 32) {
            v16bf a = load_a_frag(As + wave * 16 * GEMM_BK, GEMM_BK, k2, l16, half);
#pragma unroll
            for (int nt = 0; nt < 4; ++nt) {
                v16bf b = load_b_frag(Bs + nt * 16 * GEMM_BK, GEMM_BK, k2, l16, half);
                acc[nt] = wmma_bf16(a, b, acc[nt]);
            }
        }
        __syncthreads();
    }

#pragma unroll
    for (int nt = 0; nt < 4; ++nt) {
#pragma unroll
        for (int g = 0; g < 8; ++g) {
            int r = row0 + g + 8 * half;          // C-layout: VGPR g -> M=g (+8 for hi half)
            int c = col0 + nt * 16 + l16;         // N = lane%16
            if (OUT_BF16) ((__bf16*)Cout)[(size_t)r * N + c] = (__bf16)acc[nt][g];
            else          ((float*)Cout)[(size_t)r * N + c]  = acc[nt][g];
        }
    }
}

// ---------------------------------------------------------------------------
// Chunked RWKV scan, one block (128 threads / 4 waves) per head.
// Per chunk of C=64 steps with per-head scalar decay w=exp(-exp(td)), u=exp(tf):
//   P[t,j]   = (r_t . k_j) * (j>t ? 0 : j==t ? u : w^{t-1-j})
//   y_t      = P @ V  +  (w^t r_t) @ S0
//   S_new    = w^64 * S0 + sum_j w^{63-j} k_j v_j^T
// followed by per-(t,head) GroupNorm written as bf16 for the final GEMM.
// ---------------------------------------------------------------------------
__global__ __launch_bounds__(128) void rwkv_scan_kernel(
    const __bf16* __restrict__ Kp, const __bf16* __restrict__ Vp,
    const __bf16* __restrict__ Rp,
    const float* __restrict__ time_decay, const float* __restrict__ time_first,
    const float* __restrict__ lxw, const float* __restrict__ lxb,
    const float* __restrict__ state0,
    __bf16* __restrict__ Yn) {
    extern __shared__ char smem_raw[];
    float*  St  = (float*)smem_raw;                 // [d][s] state (transposed), f32
    float*  Yf  = St + S_DIM * S_DIM;               // [t][d] chunk output, f32
    __bf16* Stb = (__bf16*)(Yf + CHUNK * S_DIM);    // [d][s] state bf16 shadow (B operand)
    __bf16* Kb  = Stb + S_DIM * S_DIM;              // [t][s] K chunk (B operand of P)
    __bf16* Kt  = Kb + CHUNK * S_DIM;               // [s][t] w^{63-t}-scaled K^T (A of update)
    __bf16* Rb  = Kt + S_DIM * CHUNK;               // [t][s] R chunk (A operand)
    __bf16* Vt  = Rb + CHUNK * S_DIM;               // [d][t] V^T (B operand)
    __bf16* Pb  = Vt + S_DIM * CHUNK;               // [t][j] masked attention (A operand)

    const int h    = blockIdx.x;
    const int tid  = threadIdx.x;
    const int lane = tid & 31;
    const int wave = tid >> 5;
    const int half = lane >> 4;
    const int l16  = lane & 15;
    const int mt   = wave;                          // each wave owns one 16-row tile band

    const float lnw = -__expf(time_decay[h]);
    const float u   = __expf(time_first[h]);
    const float w64 = __expf(lnw * (float)CHUNK);
    const size_t colbase = (size_t)h * S_DIM;

    // state0[h][s][d] -> St[d][s]
    for (int i = tid; i < S_DIM * S_DIM; i += 128) {
        int s = i >> 6, d = i & 63;
        St[d * S_DIM + s] = state0[(size_t)h * S_DIM * S_DIM + i];
    }
    __syncthreads();

    for (int c0 = 0; c0 < T_LEN; c0 += CHUNK) {
        // ---- stage chunk into LDS (bf16, plus transposed/scaled copies) ----
        for (int i = tid; i < CHUNK * S_DIM; i += 128) {
            int t = i >> 6, s = i & 63;
            size_t g = (size_t)(c0 + t) * D_DIM + colbase + s;
            __bf16 kv = Kp[g], vv = Vp[g], rv = Rp[g];
            Kb[t * S_DIM + s] = kv;
            Kt[s * CHUNK + t] = (__bf16)((float)kv * __expf(lnw * (float)(CHUNK - 1 - t)));
            Rb[t * S_DIM + s] = rv;
            Vt[s * CHUNK + t] = vv;
        }
        for (int i = tid; i < S_DIM * S_DIM; i += 128) Stb[i] = (__bf16)St[i];
        __syncthreads();

        // ---- P = R @ K^T, then apply causal decay mask, store bf16 ----
        {
            v8f acc[4] = {};
#pragma unroll
            for (int kk = 0; kk < S_DIM; kk += 32) {
                v16bf a = load_a_frag(Rb + mt * 16 * S_DIM, S_DIM, kk, l16, half);
#pragma unroll
                for (int nt = 0; nt < 4; ++nt) {
                    v16bf b = load_b_frag(Kb + nt * 16 * S_DIM, S_DIM, kk, l16, half);
                    acc[nt] = wmma_bf16(a, b, acc[nt]);
                }
            }
#pragma unroll
            for (int nt = 0; nt < 4; ++nt) {
#pragma unroll
                for (int g = 0; g < 8; ++g) {
                    int t = mt * 16 + g + 8 * half;
                    int j = nt * 16 + l16;
                    float f = (j > t) ? 0.0f
                            : (j == t) ? u
                                       : __expf(lnw * (float)(t - 1 - j));
                    Pb[t * CHUNK + j] = (__bf16)(acc[nt][g] * f);
                }
            }
        }
        __syncthreads();

        // ---- Y = (w^t R) @ S0 + P @ V ----
        {
            v8f acc[4] = {};
            const float rscale = __expf(lnw * (float)(mt * 16 + l16)); // row t = mt*16+l16
#pragma unroll
            for (int kk = 0; kk < S_DIM; kk += 32) {
                v16bf a = load_a_frag(Rb + mt * 16 * S_DIM, S_DIM, kk, l16, half);
#pragma unroll
                for (int i = 0; i < 16; ++i) a[i] = (__bf16)((float)a[i] * rscale);
#pragma unroll
                for (int nt = 0; nt < 4; ++nt) {
                    v16bf b = load_b_frag(Stb + nt * 16 * S_DIM, S_DIM, kk, l16, half);
                    acc[nt] = wmma_bf16(a, b, acc[nt]);
                }
            }
#pragma unroll
            for (int kk = 0; kk < CHUNK; kk += 32) {
                v16bf a = load_a_frag(Pb + mt * 16 * CHUNK, CHUNK, kk, l16, half);
#pragma unroll
                for (int nt = 0; nt < 4; ++nt) {
                    v16bf b = load_b_frag(Vt + nt * 16 * CHUNK, CHUNK, kk, l16, half);
                    acc[nt] = wmma_bf16(a, b, acc[nt]);
                }
            }
#pragma unroll
            for (int nt = 0; nt < 4; ++nt)
#pragma unroll
                for (int g = 0; g < 8; ++g)
                    Yf[(mt * 16 + g + 8 * half) * S_DIM + nt * 16 + l16] = acc[nt][g];
        }
        __syncthreads();

        // ---- GroupNorm per (t, head) row; wave-uniform branch (waves 0-1) ----
        if (tid < CHUNK) {
            int t = tid;
            float s1 = 0.0f, s2 = 0.0f;
            for (int d = 0; d < S_DIM; ++d) {
                float y = Yf[t * S_DIM + d];
                s1 += y; s2 += y * y;
            }
            float mu  = s1 * (1.0f / S_DIM);
            float var = s2 * (1.0f / S_DIM) - mu * mu;
            float inv = rsqrtf(var + 1e-5f);
            for (int d = 0; d < S_DIM; ++d) {
                float y = (Yf[t * S_DIM + d] - mu) * inv * lxw[colbase + d] + lxb[colbase + d];
                Yn[(size_t)(c0 + t) * D_DIM + colbase + d] = (__bf16)y;
            }
        }

        // ---- state update: S_new = w^64 * S_old + K'^T @ V ----
        {
            v8f acc[4] = {};
#pragma unroll
            for (int kk = 0; kk < CHUNK; kk += 32) {
                v16bf a = load_a_frag(Kt + mt * 16 * CHUNK, CHUNK, kk, l16, half);
#pragma unroll
                for (int nt = 0; nt < 4; ++nt) {
                    v16bf b = load_b_frag(Vt + nt * 16 * CHUNK, CHUNK, kk, l16, half);
                    acc[nt] = wmma_bf16(a, b, acc[nt]);
                }
            }
#pragma unroll
            for (int nt = 0; nt < 4; ++nt)
#pragma unroll
                for (int g = 0; g < 8; ++g) {
                    int s = mt * 16 + g + 8 * half;
                    int d = nt * 16 + l16;
                    St[d * S_DIM + s] = w64 * St[d * S_DIM + s] + acc[nt][g];
                }
        }
        __syncthreads();
    }
}

// ---------------------------------------------------------------------------
// Launch
// ---------------------------------------------------------------------------
extern "C" void kernel_launch(void* const* d_in, const int* in_sizes, int n_in,
                              void* d_out, int out_size, void* d_ws, size_t ws_size,
                              hipStream_t stream) {
    const float* hidden     = (const float*)d_in[0];
    const float* time_decay = (const float*)d_in[1];
    const float* time_first = (const float*)d_in[2];
    const float* tmk        = (const float*)d_in[3];
    const float* tmv        = (const float*)d_in[4];
    const float* tmr        = (const float*)d_in[5];
    const float* Wk         = (const float*)d_in[6];
    const float* Wv         = (const float*)d_in[7];
    const float* Wr         = (const float*)d_in[8];
    const float* Wo         = (const float*)d_in[9];
    const float* lxw        = (const float*)d_in[10];
    const float* lxb        = (const float*)d_in[11];
    const float* state0     = (const float*)d_in[12];
    float* out = (float*)d_out;

    const size_t DD = (size_t)D_DIM * D_DIM;
    const size_t TD = (size_t)T_LEN * D_DIM;

    __bf16* p   = (__bf16*)d_ws;
    __bf16* Wkb = p; p += DD;
    __bf16* Wvb = p; p += DD;
    __bf16* Wrb = p; p += DD;
    __bf16* Wob = p; p += DD;
    __bf16* Kin = p; p += TD;
    __bf16* Vin = p; p += TD;
    __bf16* Rin = p; p += TD;
    __bf16* Kp  = p; p += TD;
    __bf16* Vp  = p; p += TD;
    __bf16* Rp  = p; p += TD;
    __bf16* Yn  = p; p += TD;   // total 11 * 8 MB = 88 MB

    // 1) weights f32 -> bf16
    {
        int n = (int)DD;
        dim3 g((n + 255) / 256), b(256);
        cvt_f32_bf16_kernel<<<g, b, 0, stream>>>(Wk, Wkb, n);
        cvt_f32_bf16_kernel<<<g, b, 0, stream>>>(Wv, Wvb, n);
        cvt_f32_bf16_kernel<<<g, b, 0, stream>>>(Wr, Wrb, n);
        cvt_f32_bf16_kernel<<<g, b, 0, stream>>>(Wo, Wob, n);
    }
    // 2) time-shift mix
    {
        int n = (int)TD;
        time_mix_kernel<<<(n + 255) / 256, 256, 0, stream>>>(
            hidden, tmk, tmv, tmr, Kin, Vin, Rin);
    }
    // 3) projections (bf16 WMMA + async LDS staging, bf16 out)
    {
        dim3 g(T_LEN / GEMM_BM, D_DIM / GEMM_BN), b(256);
        gemm_wmma_kernel<true><<<g, b, 0, stream>>>(Kin, Wkb, Kp, T_LEN, D_DIM, D_DIM);
        gemm_wmma_kernel<true><<<g, b, 0, stream>>>(Vin, Wvb, Vp, T_LEN, D_DIM, D_DIM);
        gemm_wmma_kernel<true><<<g, b, 0, stream>>>(Rin, Wrb, Rp, T_LEN, D_DIM, D_DIM);
    }
    // 4) chunked scan + GroupNorm (one block per head, 80 KB dynamic LDS)
    {
        size_t smem = (size_t)(S_DIM * S_DIM + CHUNK * S_DIM) * sizeof(float)
                    + (size_t)(S_DIM * S_DIM + 5 * CHUNK * S_DIM) * sizeof(__bf16);
        rwkv_scan_kernel<<<H_NUM, 128, smem, stream>>>(
            Kp, Vp, Rp, time_decay, time_first, lxw, lxb, state0, Yn);
    }
    // 5) output projection (f32 out)
    {
        dim3 g(T_LEN / GEMM_BM, D_DIM / GEMM_BN), b(256);
        gemm_wmma_kernel<false><<<g, b, 0, stream>>>(Yn, Wob, out, T_LEN, D_DIM, D_DIM);
    }
}
